// TFFI_1537598292584
// MI455X (gfx1250) — matrix operand order repaired and checked by reading, not compiled, not run on hardware
//
#include <hip/hip_runtime.h>
#include <cstdint>

// ---------------------------------------------------------------------------
// TFSA block for MI455X (gfx1250, wave32, WMMA 16x16x32 f16 -> f32 acc)
//
//   0) cvt_kernel  : one-time fp32->fp16 weight conversion; folds BN scale
//                    into the projection weight matrix, precomputes folded
//                    bias / PReLU alpha.
//   1) proj_kernel : conv1x1 (64->80) as WMMA GEMM per 128-pixel tile; BN
//                    bias + PReLU + channel de-interleave in epilogue;
//                    softmax scale folded into Q.
//   2) fattn_kernel: per (b,t) 256x256 attention over frequency (head dim 16)
//   3) tattn_kernel: per (b,f) 256x256 CAUSAL attention over time, V = f_out
//   4) post_kernel : wp conv + BN + PReLU + residual, reshape-semantics LN
//                    over contiguous 64-runs of T, 64->384->64 MLP (WMMA,
//                    exact GELU), gamma, transpose-scatter + input residual.
//
// All matrix fragments are built from zero-padded, conflict-padded LDS tiles
// with unconditional contiguous 16B loads (no divergent guarded ds loads).
// Workspace: 7 x 8MB fp16 intermediates + fp16 weights + folded params.
// ---------------------------------------------------------------------------

typedef __attribute__((ext_vector_type(16))) _Float16 v16h;
typedef __attribute__((ext_vector_type(8)))  _Float16 v8h;
typedef __attribute__((ext_vector_type(8)))  float    v8f;

#define B_  4
#define C_  64
#define F_  256
#define T_  256
#define DC_ 16

#define QSTR 40    // Q/K LDS row stride (halfs): 80B, 16B-aligned, bank-spread
#define PSTR 40    // P transpose buffer row stride
#define VSTR 264   // transposed-V row stride: 528B
#define ZSTR 72    // X/Z LDS row stride: 144B
#define HSTR 200   // Hs row stride: 400B

__device__ __forceinline__ v16h cat8(v8h a, v8h b) {
  return __builtin_shufflevector(a, b, 0, 1, 2, 3, 4, 5, 6, 7,
                                 8, 9, 10, 11, 12, 13, 14, 15);
}

__device__ __forceinline__ v8f zero8() {
  v8f z = {0.f, 0.f, 0.f, 0.f, 0.f, 0.f, 0.f, 0.f};
  return z;
}

__device__ __forceinline__ v8f wmma16(v16h a, v16h b, v8f c) {
  return __builtin_amdgcn_wmma_f32_16x16x32_f16(false, a, false, b, (short)0, c,
                                                false, false);
}

// ---------------------------------------------------------------------------
// Kernel 0: one-time weight prep.
//   w1h/w2h : fp16 copies of the MLP weights (row-major, B-fragment friendly)
//   Wh      : [80][64] fp16 projection weights with BN scale folded in
//   biasW   : [80] folded BN bias,  alphaW: [80] PReLU alpha
// ---------------------------------------------------------------------------
__global__ __launch_bounds__(256) void cvt_kernel(
    const float* __restrict__ w1, const float* __restrict__ w2,
    const float* __restrict__ wf, const float* __restrict__ fg,
    const float* __restrict__ fb, const float* __restrict__ fm,
    const float* __restrict__ fv, const float* __restrict__ fa,
    const float* __restrict__ wt, const float* __restrict__ tg,
    const float* __restrict__ tb, const float* __restrict__ tm,
    const float* __restrict__ tv, const float* __restrict__ ta,
    _Float16* __restrict__ w1h, _Float16* __restrict__ w2h,
    _Float16* __restrict__ Wh, float* __restrict__ biasW,
    float* __restrict__ alphaW)
{
  int i = blockIdx.x * 256 + threadIdx.x;
  if (i < 384 * 64) {
    w1h[i] = (_Float16)w1[i];
    w2h[i] = (_Float16)w2[i];
  }
  if (i < 80 * 64) {
    int o = i >> 6, cc = i & 63;
    float w, inv;
    if (o < 48) { inv = fg[o] * rsqrtf(fv[o] + 1e-5f); w = wf[o * 64 + cc]; }
    else { int o2 = o - 48; inv = tg[o2] * rsqrtf(tv[o2] + 1e-5f); w = wt[o2 * 64 + cc]; }
    Wh[i] = (_Float16)(w * inv);
  }
  if (i < 80) {
    if (i < 48) {
      float inv = fg[i] * rsqrtf(fv[i] + 1e-5f);
      biasW[i] = fb[i] - fm[i] * inv; alphaW[i] = fa[i];
    } else {
      int o2 = i - 48;
      float inv = tg[o2] * rsqrtf(tv[o2] + 1e-5f);
      biasW[i] = tb[o2] - tm[o2] * inv; alphaW[i] = ta[o2];
    }
  }
}

// ---------------------------------------------------------------------------
// Kernel 1: projection as WMMA GEMM.  Block = 128 pixels, M=128, N=80, K=64.
// ---------------------------------------------------------------------------
__global__ __launch_bounds__(256) void proj_kernel(
    const float* __restrict__ x, const _Float16* __restrict__ Wh,
    const float* __restrict__ biasW, const float* __restrict__ alphaW,
    _Float16* __restrict__ Qf, _Float16* __restrict__ Kf,
    _Float16* __restrict__ Vf, _Float16* __restrict__ Qt,
    _Float16* __restrict__ Kt)
{
  __shared__ __align__(16) _Float16 Xs[128 * ZSTR];   // X^T tile (M x K) f16
  __shared__ __align__(16) float    Os[128 * 80];     // raw conv outputs
  __shared__ float bS[80], aS[80];

  int tid = threadIdx.x;
  int p0  = blockIdx.x * 128;            // 128 consecutive pixels (same b)
  int b   = p0 >> 16;

  const float* xb = x + (size_t)b * (C_ * 65536) + (p0 & 65535);
  for (int i = tid; i < 64 * 128; i += 256) {
    int cc = i >> 7, pl = i & 127;       // coalesced global read along pixels
    Xs[pl * ZSTR + cc] = (_Float16)xb[(size_t)cc * 65536 + pl];
  }
  if (tid < 80) { bS[tid] = biasW[tid]; aS[tid] = alphaW[tid]; }
  __syncthreads();

  int lane = tid & 31, wave = tid >> 5;
  int g = lane >> 4, ln = lane & 15;

  v16h ax[2];
#pragma unroll
  for (int ks = 0; ks < 2; ks++) {
    const _Float16* p = &Xs[(wave * 16 + ln) * ZSTR + ks * 32 + (g << 3)];
    ax[ks] = cat8(*(const v8h*)p, *(const v8h*)(p + 16));
  }
#pragma unroll
  for (int nt = 0; nt < 5; nt++) {       // 5 N-tiles = 80 output channels
    v8f cacc = zero8();
#pragma unroll
    for (int ks = 0; ks < 2; ks++) {
      const _Float16* pw = &Wh[(size_t)(nt * 16 + ln) * 64 + ks * 32 + (g << 3)];
      v16h bw = cat8(*(const v8h*)pw, *(const v8h*)(pw + 16));
      cacc = wmma16(ax[ks], bw, cacc);
    }
#pragma unroll
    for (int r = 0; r < 8; r++)
      Os[(wave * 16 + r + (g << 3)) * 80 + nt * 16 + ln] = cacc[r];
  }
  __syncthreads();

  // epilogue: bias + PReLU + de-interleave, uniform control flow per thread
  int pl = tid >> 1, h = tid & 1;
  int p = p0 + pl;
  int f = (p >> 8) & 255, t = p & 255;
  const float SC = 0.25f;                // 1/sqrt(DC), folded into Q
  if (h == 0) {
    size_t fbase = (((size_t)b * T_ + t) * F_ + f) * DC_;   // [B][T][F][DC]
#pragma unroll
    for (int o = 0; o < 48; o++) {
      float v = Os[pl * 80 + o] + bS[o];
      v = (v >= 0.f) ? v : aS[o] * v;
      int j = o / 3, wch = o - 3 * j;
      if (wch == 0)      Qf[fbase + j] = (_Float16)(v * SC);
      else if (wch == 1) Kf[fbase + j] = (_Float16)v;
      else               Vf[fbase + j] = (_Float16)v;
    }
  } else {
    size_t tbase = (size_t)p * DC_;      // [B][F][T][DC] is pixel-linear
#pragma unroll
    for (int o = 0; o < 32; o++) {
      float v = Os[pl * 80 + 48 + o] + bS[48 + o];
      v = (v >= 0.f) ? v : aS[48 + o] * v;
      int j = o >> 1;
      if ((o & 1) == 0) Qt[tbase + j] = (_Float16)(v * SC);
      else              Kt[tbase + j] = (_Float16)v;
    }
  }
}

// ---------------------------------------------------------------------------
// Shared attention helpers: stage Q/K zero-padded to 32 K-columns, V
// transposed, so every fragment is two contiguous 16B LDS loads.
// ---------------------------------------------------------------------------
__device__ __forceinline__ void stage_qkv(
    const _Float16* __restrict__ Qg, const _Float16* __restrict__ Kg,
    const _Float16* __restrict__ Vg, size_t base,
    _Float16* Qs, _Float16* Ks, _Float16* Vt, int tid)
{
  const uint4* q4 = (const uint4*)(Qg + base);
  const uint4* k4 = (const uint4*)(Kg + base);
  const v8h*   v8 = (const v8h*)(Vg + base);
  uint4 zz = {0u, 0u, 0u, 0u};
  for (int i = tid; i < 512; i += 256) {
    int row = i >> 1, part = (i & 1) << 3;
    *(uint4*)&Qs[row * QSTR + part]      = q4[i];
    *(uint4*)&Ks[row * QSTR + part]      = k4[i];
    *(uint4*)&Qs[row * QSTR + 16 + part] = zz;
    *(uint4*)&Ks[row * QSTR + 16 + part] = zz;
    int y = i >> 1, c0 = (i & 1) << 3;
    v8h vv = v8[i];
#pragma unroll
    for (int u = 0; u < 8; u++) Vt[(c0 + u) * VSTR + y] = vv[u];
  }
}

__device__ __forceinline__ v16h frag_a(const _Float16* P, int row, int stride,
                                       int g) {
  const _Float16* p = P + row * stride + (g << 3);
  return cat8(*(const v8h*)p, *(const v8h*)(p + 16));
}

__device__ __forceinline__ v16h frag_b_row(const _Float16* P, int row,
                                           int stride, int g) {
  const _Float16* p = P + row * stride + (g << 3);
  return cat8(*(const v8h*)p, *(const v8h*)(p + 16));
}

// ---------------------------------------------------------------------------
// Kernel 2: frequency attention, one block per (b, t).
// ---------------------------------------------------------------------------
__global__ __launch_bounds__(256) void fattn_kernel(
    const _Float16* __restrict__ Qf, const _Float16* __restrict__ Kf,
    const _Float16* __restrict__ Vf, _Float16* __restrict__ Fout)
{
  __shared__ __align__(16) _Float16 Qs[F_ * QSTR];
  __shared__ __align__(16) _Float16 Ks[F_ * QSTR];
  __shared__ __align__(16) _Float16 Vt[DC_ * VSTR];
  __shared__ __align__(16) _Float16 Pws[8][16 * PSTR];

  int tid = threadIdx.x;
  int bt  = blockIdx.x;                  // b*T + t
  size_t base = (size_t)bt * (F_ * DC_);
  stage_qkv(Qf, Kf, Vf, base, Qs, Ks, Vt, tid);
  __syncthreads();

  int lane = tid & 31, wave = tid >> 5;
  int g = lane >> 4, ln = lane & 15;
  int b = bt >> 8, t = bt & 255;

  for (int qi = 0; qi < 2; qi++) {
    int qt_ = wave * 2 + qi;             // query tile (16 rows)
    v16h aq = frag_a(Qs, qt_ * 16 + ln, QSTR, g);

    // ---- pass 1: row maxes ----
    float vmax[8];
#pragma unroll
    for (int r = 0; r < 8; r++) vmax[r] = -3.0e38f;
    for (int kt = 0; kt < 16; kt++) {
      v16h bk = frag_b_row(Ks, kt * 16 + ln, QSTR, g);
      v8f s = wmma16(aq, bk, zero8());
#pragma unroll
      for (int r = 0; r < 8; r++) vmax[r] = fmaxf(vmax[r], s[r]);
    }
#pragma unroll
    for (int off = 1; off < 16; off <<= 1)
#pragma unroll
      for (int r = 0; r < 8; r++)
        vmax[r] = fmaxf(vmax[r], __shfl_xor(vmax[r], off, 32));

    // ---- pass 2: exp, sums, O = P*V ----
    float vsum[8];
#pragma unroll
    for (int r = 0; r < 8; r++) vsum[r] = 0.f;
    v8f accO = zero8();
    _Float16* Pw = &Pws[wave][0];

    for (int kc = 0; kc < 8; kc++) {     // 32 key columns per chunk
#pragma unroll
      for (int sub = 0; sub < 2; sub++) {
        int kt = kc * 2 + sub;
        v16h bk = frag_b_row(Ks, kt * 16 + ln, QSTR, g);
        v8f s = wmma16(aq, bk, zero8());
#pragma unroll
        for (int r = 0; r < 8; r++) {
          float pv = __expf(s[r] - vmax[r]);
          vsum[r] += pv;
          Pw[(r + (g << 3)) * PSTR + sub * 16 + ln] = (_Float16)pv;
        }
      }
      v16h ap = frag_a(Pw, ln, PSTR, g);                 // P as A-fragment
      v16h bv = frag_b_row(Vt + kc * 32, ln, VSTR, g);   // V^T rows
      accO = wmma16(ap, bv, accO);
    }
#pragma unroll
    for (int off = 1; off < 16; off <<= 1)
#pragma unroll
      for (int r = 0; r < 8; r++) vsum[r] += __shfl_xor(vsum[r], off, 32);

#pragma unroll
    for (int r = 0; r < 8; r++) {
      int frow = qt_ * 16 + r + (g << 3);
      float o = accO[r] / vsum[r];
      Fout[(((size_t)b * F_ + frow) * T_ + t) * DC_ + ln] = (_Float16)o;
    }
  }
}

// ---------------------------------------------------------------------------
// Kernel 3: causal time attention, one block per (b, f).  V = f_out.
// ---------------------------------------------------------------------------
__global__ __launch_bounds__(256) void tattn_kernel(
    const _Float16* __restrict__ Qt, const _Float16* __restrict__ Kt,
    const _Float16* __restrict__ Fout, _Float16* __restrict__ Tout)
{
  __shared__ __align__(16) _Float16 Qs[T_ * QSTR];
  __shared__ __align__(16) _Float16 Ks[T_ * QSTR];
  __shared__ __align__(16) _Float16 Vt[DC_ * VSTR];
  __shared__ __align__(16) _Float16 Pws[8][16 * PSTR];

  int tid = threadIdx.x;
  int bf  = blockIdx.x;                  // b*F + f
  size_t base = (size_t)bf * (T_ * DC_);
  stage_qkv(Qt, Kt, Fout, base, Qs, Ks, Vt, tid);
  __syncthreads();

  int lane = tid & 31, wave = tid >> 5;
  int g = lane >> 4, ln = lane & 15;

  for (int qi = 0; qi < 2; qi++) {
    int qt_ = wave * 2 + qi;
    v16h aq = frag_a(Qs, qt_ * 16 + ln, QSTR, g);

    float vmax[8];
#pragma unroll
    for (int r = 0; r < 8; r++) vmax[r] = -3.0e38f;
    for (int kt = 0; kt <= qt_; kt++) {  // causal: tiles above diagonal skipped
      v16h bk = frag_b_row(Ks, kt * 16 + ln, QSTR, g);
      v8f s = wmma16(aq, bk, zero8());
#pragma unroll
      for (int r = 0; r < 8; r++) {
        int trow = qt_ * 16 + r + (g << 3);
        int y    = kt * 16 + ln;
        if (y <= trow) vmax[r] = fmaxf(vmax[r], s[r]);
      }
    }
#pragma unroll
    for (int off = 1; off < 16; off <<= 1)
#pragma unroll
      for (int r = 0; r < 8; r++)
        vmax[r] = fmaxf(vmax[r], __shfl_xor(vmax[r], off, 32));

    float vsum[8];
#pragma unroll
    for (int r = 0; r < 8; r++) vsum[r] = 0.f;
    v8f accO = zero8();
    _Float16* Pw = &Pws[wave][0];
    int kcmax = qt_ >> 1;

    for (int kc = 0; kc <= kcmax; kc++) {
#pragma unroll
      for (int sub = 0; sub < 2; sub++) {
        int kt = kc * 2 + sub;
        if (kt <= qt_) {                 // wave-uniform branch
          v16h bk = frag_b_row(Ks, kt * 16 + ln, QSTR, g);
          v8f s = wmma16(aq, bk, zero8());
#pragma unroll
          for (int r = 0; r < 8; r++) {
            int trow = qt_ * 16 + r + (g << 3);
            int y    = kt * 16 + ln;
            float pv = (y <= trow) ? __expf(s[r] - vmax[r]) : 0.f;
            vsum[r] += pv;
            Pw[(r + (g << 3)) * PSTR + sub * 16 + ln] = (_Float16)pv;
          }
        } else {
#pragma unroll
          for (int r = 0; r < 8; r++)
            Pw[(r + (g << 3)) * PSTR + sub * 16 + ln] = (_Float16)0.f;
        }
      }
      v16h ap = frag_a(Pw, ln, PSTR, g);
      v16h bv = frag_b_row(Vt + kc * 32, ln, VSTR, g);
      accO = wmma16(ap, bv, accO);
    }
#pragma unroll
    for (int off = 1; off < 16; off <<= 1)
#pragma unroll
      for (int r = 0; r < 8; r++) vsum[r] += __shfl_xor(vsum[r], off, 32);

#pragma unroll
    for (int r = 0; r < 8; r++) {
      int trow = qt_ * 16 + r + (g << 3);
      float o = accO[r] / vsum[r];
      Tout[((size_t)bf * T_ + trow) * DC_ + ln] = (_Float16)o;
    }
  }
}

// ---------------------------------------------------------------------------
// Kernel 4: wp conv + BN + PReLU + residual, reshape-semantics LN + MLP,
// gamma scale, transpose scatter + input residual.
// One block per (b, f, t0), t0 in {0,64,128,192}: 64 runs x 64 channels.
// ---------------------------------------------------------------------------
__global__ __launch_bounds__(256) void post_kernel(
    const float* __restrict__ x, const _Float16* __restrict__ Tout,
    const float* __restrict__ wp, const float* __restrict__ pg,
    const float* __restrict__ pb, const float* __restrict__ pm,
    const float* __restrict__ pv, const float* __restrict__ pa,
    const float* __restrict__ lnw, const float* __restrict__ lnb,
    const _Float16* __restrict__ w1h, const float* __restrict__ b1,
    const _Float16* __restrict__ w2h, const float* __restrict__ b2,
    const float* __restrict__ gamma, float* __restrict__ out)
{
  __shared__ __align__(16) _Float16 ToutS[64 * 16];
  __shared__ __align__(16) float    WpS[64 * 16];
  __shared__ float PsS[64], PbS[64], PaS[64], LwS[64], LbS[64];
  __shared__ __align__(16) _Float16 Zs[64 * ZSTR];
  __shared__ __align__(16) _Float16 Hs[64 * HSTR];

  int tid = threadIdx.x;
  int blk = blockIdx.x;
  int bf  = blk >> 2;
  int t0  = (blk & 3) << 6;
  int b   = bf >> 8, f = bf & 255;

  if (tid < 128)
    ((uint4*)ToutS)[tid] =
        ((const uint4*)(Tout + ((size_t)bf * T_ + t0) * DC_))[tid];
  for (int i = tid; i < 64 * 16; i += 256) WpS[i] = wp[i];
  if (tid < 64) {
    float inv = pg[tid] * rsqrtf(pv[tid] + 1e-5f);
    PsS[tid] = inv; PbS[tid] = pb[tid] - pm[tid] * inv; PaS[tid] = pa[tid];
    LwS[tid] = lnw[tid]; LbS[tid] = lnb[tid];
  }
  __syncthreads();

  // ---- stage 1: y = prelu(bn(wp . tout)) + x, then LN over each 64-run ----
  int c = tid >> 2, q = tid & 3;         // 4 threads per run (channel c)
  float wrow[16];
#pragma unroll
  for (int j = 0; j < 16; j++) wrow[j] = WpS[c * 16 + j];

  float z[16], s1 = 0.f, s2 = 0.f;
  const float* xrow =
      x + ((size_t)(b * C_ + c)) * 65536 + f * 256 + t0 + q * 16;
#pragma unroll
  for (int i = 0; i < 16; i++) {
    int tt = q * 16 + i;
    v8h t0v = *(const v8h*)&ToutS[tt * 16];
    v8h t1v = *(const v8h*)&ToutS[tt * 16 + 8];
    float a = 0.f;
#pragma unroll
    for (int j = 0; j < 8; j++) a += wrow[j] * (float)t0v[j];
#pragma unroll
    for (int j = 0; j < 8; j++) a += wrow[8 + j] * (float)t1v[j];
    a = a * PsS[c] + PbS[c];
    a = (a >= 0.f) ? a : PaS[c] * a;
    a += xrow[i];
    z[i] = a; s1 += a; s2 += a * a;
  }
  s1 += __shfl_xor(s1, 1, 32); s1 += __shfl_xor(s1, 2, 32);
  s2 += __shfl_xor(s2, 1, 32); s2 += __shfl_xor(s2, 2, 32);
  float mu  = s1 * 0.015625f;
  float var = s2 * 0.015625f - mu * mu;
  float rs  = rsqrtf(var + 1e-6f);
#pragma unroll
  for (int i = 0; i < 16; i++) {
    int tt = q * 16 + i;                 // LN channel = position in run
    Zs[c * ZSTR + tt] = (_Float16)((z[i] - mu) * rs * LwS[tt] + LbS[tt]);
  }
  __syncthreads();

  // ---- stage 2: MLP 64 -> 384 (gelu) -> 64, WMMA, K-chunked over 2 halves --
  int lane = tid & 31, wave = tid >> 5;
  int g = lane >> 4, ln = lane & 15;
  int mt = wave >> 1, nsub = wave & 1;   // 2 waves share an M-tile, split N

  v16h az[2];
#pragma unroll
  for (int ks = 0; ks < 2; ks++) {
    const _Float16* p = &Zs[(mt * 16 + ln) * ZSTR + ks * 32 + (g << 3)];
    az[ks] = cat8(*(const v8h*)p, *(const v8h*)(p + 16));
  }

  v8f acco[2];
  acco[0] = zero8(); acco[1] = zero8();

  for (int nh = 0; nh < 2; nh++) {       // hidden halves of 192 columns
    // GEMM1: H[mt rows][this wave's 6 tiles] = gelu(Z*W1^T + b1)
    for (int ntl = 0; ntl < 6; ntl++) {
      int colb = (nsub * 6 + ntl) * 16;  // column base within half
      int col  = nh * 192 + colb + ln;   // this lane's global hidden column
      v8f ch = zero8();
#pragma unroll
      for (int ks = 0; ks < 2; ks++) {
        const _Float16* p = &w1h[(size_t)col * 64 + ks * 32 + (g << 3)];
        v16h bw = cat8(*(const v8h*)p, *(const v8h*)(p + 16));
        ch = wmma16(az[ks], bw, ch);
      }
      float bias = b1[col];
#pragma unroll
      for (int r = 0; r < 8; r++) {
        float hv = ch[r] + bias;
        hv = 0.5f * hv * (1.f + erff(hv * 0.70710678f));   // exact gelu
        Hs[(mt * 16 + r + (g << 3)) * HSTR + colb + ln] = (_Float16)hv;
      }
    }
    __syncthreads();
    // GEMM2 partial: acc += H_half * W2^T (this wave: its M-tile, 2 N-tiles)
    for (int ks2 = 0; ks2 < 6; ks2++) {
      const _Float16* ph = &Hs[(mt * 16 + ln) * HSTR + ks2 * 32 + (g << 3)];
      v16h ah = cat8(*(const v8h*)ph, *(const v8h*)(ph + 16));
#pragma unroll
      for (int nt2 = 0; nt2 < 2; nt2++) {
        int nch = nsub * 32 + nt2 * 16 + ln;
        const _Float16* pw =
            &w2h[(size_t)nch * 384 + nh * 192 + ks2 * 32 + (g << 3)];
        v16h bw2 = cat8(*(const v8h*)pw, *(const v8h*)(pw + 16));
        acco[nt2] = wmma16(ah, bw2, acco[nt2]);
      }
    }
    __syncthreads();
  }

  // ---- final: out[b][n][c*4 + f/64][(f&63)*4 + t0/64] = x + gamma*(y+b2) ---
  int hw = ((f >> 6) << 8) + ((f & 63) << 2) + (t0 >> 6);
  size_t bbase = (size_t)b * (C_ * 65536);
#pragma unroll
  for (int nt2 = 0; nt2 < 2; nt2++) {
    int n = nsub * 32 + nt2 * 16 + ln;
    float gm = gamma[n], bb = b2[n];
#pragma unroll
    for (int r = 0; r < 8; r++) {
      int m = mt * 16 + r + (g << 3);    // run channel c
      size_t addr = bbase + (size_t)n * 65536 + (size_t)m * 1024 + hw;
      out[addr] = x[addr] + gm * (acco[nt2][r] + bb);
    }
  }
}

// ---------------------------------------------------------------------------
extern "C" void kernel_launch(void* const* d_in, const int* in_sizes, int n_in,
                              void* d_out, int out_size, void* d_ws,
                              size_t ws_size, hipStream_t stream)
{
  (void)in_sizes; (void)n_in; (void)out_size; (void)ws_size;

  const float* x    = (const float*)d_in[0];
  const float* wf   = (const float*)d_in[1];
  const float* f_g  = (const float*)d_in[2];
  const float* f_b  = (const float*)d_in[3];
  const float* f_m  = (const float*)d_in[4];
  const float* f_v  = (const float*)d_in[5];
  const float* f_a  = (const float*)d_in[6];
  const float* wt   = (const float*)d_in[7];
  const float* t_g  = (const float*)d_in[8];
  const float* t_b  = (const float*)d_in[9];
  const float* t_m  = (const float*)d_in[10];
  const float* t_v  = (const float*)d_in[11];
  const float* t_a  = (const float*)d_in[12];
  const float* wp   = (const float*)d_in[13];
  const float* p_g  = (const float*)d_in[14];
  const float* p_b  = (const float*)d_in[15];
  const float* p_m  = (const float*)d_in[16];
  const float* p_v  = (const float*)d_in[17];
  const float* p_a  = (const float*)d_in[18];
  const float* ln_w = (const float*)d_in[19];
  const float* ln_b = (const float*)d_in[20];
  const float* w1   = (const float*)d_in[21];
  const float* b1   = (const float*)d_in[22];
  const float* w2   = (const float*)d_in[23];
  const float* b2   = (const float*)d_in[24];
  const float* gmm  = (const float*)d_in[25];
  float* out = (float*)d_out;

  const size_t NPIX = (size_t)B_ * F_ * T_;   // 262144
  const size_t SEG  = NPIX * DC_;             // halfs per intermediate buffer
  _Float16* ws = (_Float16*)d_ws;             // needs 7*SEG*2 + ~110KB
  _Float16* Qf  = ws;
  _Float16* Kf  = Qf + SEG;
  _Float16* Vf  = Kf + SEG;
  _Float16* Qt  = Vf + SEG;
  _Float16* Kt  = Qt + SEG;
  _Float16* Fo  = Kt + SEG;
  _Float16* To  = Fo + SEG;
  _Float16* w1h = To + SEG;
  _Float16* w2h = w1h + 384 * 64;
  _Float16* Whh = w2h + 384 * 64;
  float* biasW  = (float*)(Whh + 80 * 64);    // 4B-aligned (even half count)
  float* alphaW = biasW + 80;

  cvt_kernel<<<96, 256, 0, stream>>>(w1, w2, wf, f_g, f_b, f_m, f_v, f_a,
                                     wt, t_g, t_b, t_m, t_v, t_a,
                                     w1h, w2h, Whh, biasW, alphaW);
  proj_kernel<<<(int)(NPIX / 128), 256, 0, stream>>>(
      x, Whh, biasW, alphaW, Qf, Kf, Vf, Qt, Kt);
  fattn_kernel<<<B_ * T_, 256, 0, stream>>>(Qf, Kf, Vf, Fo);
  tattn_kernel<<<B_ * F_, 256, 0, stream>>>(Qt, Kt, Fo, To);
  post_kernel<<<B_ * F_ * (T_ / 64), 256, 0, stream>>>(
      x, To, wp, p_g, p_b, p_m, p_v, p_a, ln_w, ln_b, w1h, b1, w2h, b2, gmm,
      out);
}
